// SelfAttentionV1_71871982731820
// MI455X (gfx1250) — matrix operand order repaired and checked by reading, not compiled, not run on hardware
//
#include <hip/hip_runtime.h>

typedef __attribute__((ext_vector_type(16))) _Float16 v16h;
typedef __attribute__((ext_vector_type(8)))  _Float16 v8h;
typedef __attribute__((ext_vector_type(8)))  float    v8f;
typedef __attribute__((ext_vector_type(4)))  float    v4f;

constexpr int H  = 512;
constexpr int B  = 4;
constexpr int S  = 4096;
constexpr int MT = B * S;           // 16384 total rows
constexpr int KEY_BLK = 1024;       // keys per online-softmax megablock
constexpr int NMB = S / KEY_BLK;    // 4 megablocks

union FragH { v16h v; v8h h[2]; _Float16 e[16]; };
union Acc   { v8f  v; float f[8]; };

__device__ __forceinline__ v8f wmma_f16(const v16h& a, const v16h& b, const v8f& c) {
    // D = A(16x32 f16) x B(32x16 f16) + C(16x16 f32)
    return __builtin_amdgcn_wmma_f32_16x16x32_f16(false, a, false, b, (short)0, c,
                                                  false, false);
}

// ---------------------------------------------------------------------------
// Kernel 0: convert W[in][out] f32 -> Wt[out][in] f16 (col-major) for 3 mats
// ---------------------------------------------------------------------------
__global__ void wt_kernel(const float* __restrict__ Wq, const float* __restrict__ Wk,
                          const float* __restrict__ Wv, _Float16* __restrict__ Wt) {
    int idx = blockIdx.x * blockDim.x + threadIdx.x;      // [0, 3*H*H)
    int mat = idx / (H * H);
    int r   = (idx / H) % H;    // input dim (k)
    int c   = idx % H;          // output dim (n)
    const float* W = (mat == 0) ? Wq : (mat == 1 ? Wk : Wv);
    Wt[(size_t)mat * H * H + (size_t)c * H + r] = (_Float16)W[(size_t)r * H + c];
}

// ---------------------------------------------------------------------------
// Kernel 1: QKV projection GEMM. Each wave: 16(M) x 64(N) tile, K-loop of 32.
// Q gets 1/sqrt(H) folded in; V is written transposed Vt[b][d][s].
// ---------------------------------------------------------------------------
__global__ void __launch_bounds__(256)
proj_kernel(const float* __restrict__ X, const _Float16* __restrict__ Wt,
            const float* __restrict__ bq, const float* __restrict__ bk,
            const float* __restrict__ bv,
            _Float16* __restrict__ Qh, _Float16* __restrict__ Kh,
            _Float16* __restrict__ Vt) {
    const int lane  = threadIdx.x & 31;
    const int ln16  = lane & 15;
    const int hi    = lane >> 4;                       // lane half: 0 or 1
    const int wid   = blockIdx.x * 8 + (threadIdx.x >> 5);
    const int mat   = wid >> 13;                       // 8192 waves per matrix
    const int rem   = wid & 8191;
    const int mtile = rem >> 3;                        // [0,1024): M/16
    const int nb    = (rem & 7) * 64;                  // N base (64-wide slab)
    const int row   = mtile * 16 + ln16;               // global M row (A layout)
    const float* bias = (mat == 0) ? bq : (mat == 1 ? bk : bv);
    const _Float16* W = Wt + (size_t)mat * H * H;

    Acc acc[4];
#pragma unroll
    for (int t = 0; t < 4; ++t)
#pragma unroll
        for (int r = 0; r < 8; ++r) acc[t].f[r] = 0.0f;

    for (int kc = 0; kc < H / 32; ++kc) {
        // A fragment: X row, halves 0..7 at k0, 8..15 at k0+16 (ISA 16b A map)
        const int k0 = kc * 32 + (hi ? 8 : 0);
        const float* xp = X + (size_t)row * H + k0;
        v4f x0 = *(const v4f*)(xp);
        v4f x1 = *(const v4f*)(xp + 4);
        v4f x2 = *(const v4f*)(xp + 16);
        v4f x3 = *(const v4f*)(xp + 20);
        FragH a;
#pragma unroll
        for (int i = 0; i < 4; ++i) {
            a.e[i]      = (_Float16)x0[i];
            a.e[4 + i]  = (_Float16)x1[i];
            a.e[8 + i]  = (_Float16)x2[i];
            a.e[12 + i] = (_Float16)x3[i];
        }
        // B fragment: lane holds column n, K contiguous (col-major Wt)
        const int kb = kc * 32 + (hi ? 8 : 0);
        const _Float16* wp0 = W + (size_t)(nb + ln16) * H + kb;
#pragma unroll
        for (int nt = 0; nt < 4; ++nt) {
            const _Float16* wp = wp0 + (size_t)nt * 16 * H;
            FragH bf;
            bf.h[0] = *(const v8h*)(wp);
            bf.h[1] = *(const v8h*)(wp + 16);
            acc[nt].v = wmma_f16(a.v, bf.v, acc[nt].v);
        }
    }

    const float qscale = 0.044194173824159216f;  // 1/sqrt(512)
#pragma unroll
    for (int nt = 0; nt < 4; ++nt) {
        const int ncol = nb + nt * 16 + ln16;
        const float bb = bias[ncol];
#pragma unroll
        for (int r = 0; r < 8; ++r) {
            const int m = mtile * 16 + r + hi * 8;   // C/D layout row
            float val = acc[nt].f[r] + bb;
            if (mat == 0) {
                Qh[(size_t)m * H + ncol] = (_Float16)(val * qscale);
            } else if (mat == 1) {
                Kh[(size_t)m * H + ncol] = (_Float16)val;
            } else {
                const int bidx = m >> 12;            // m / S
                const int sidx = m & (S - 1);
                Vt[((size_t)bidx * H + ncol) * S + sidx] = (_Float16)val;
            }
        }
    }
}

// ---------------------------------------------------------------------------
// Kernel 2: block-flash attention. 128 threads (4 waves) per 16 queries.
// Keys in 1024-wide megablocks; scores staged in LDS f16; online rescale.
// Phase A: each wave computes 16x64 scores per pass (A-frag reused 4x).
// Wave w owns hidden columns [128w, 128w+128) of the output in phase C.
// ---------------------------------------------------------------------------
__global__ void __launch_bounds__(128)
attn_kernel(const _Float16* __restrict__ Qh, const _Float16* __restrict__ Kh,
            const _Float16* __restrict__ Vt, float* __restrict__ out) {
    __shared__ __align__(16) _Float16 sQ[16 * H];         // 16 KB
    __shared__ __align__(16) _Float16 sP[16 * KEY_BLK];   // 32 KB
    __shared__ float sRowMax[16];
    __shared__ float sRowSum[16];
    __shared__ float sFactor[16];
    __shared__ float sPMax[128][8];
    __shared__ float sPSum[128];

    const int tid  = threadIdx.x;
    const int lane = tid & 31;
    const int ln16 = lane & 15;
    const int hi   = lane >> 4;
    const int w    = tid >> 5;                 // wave 0..3
    const int b    = blockIdx.x >> 8;          // / 256
    const int q0   = (blockIdx.x & 255) * 16;

    // Stage Q tile (16 x 512 f16) into LDS, coalesced
    {
        const _Float16* qsrc = Qh + (size_t)(b * S + q0) * H;
#pragma unroll
        for (int i = 0; i < 8; ++i) {
            int e = (i * 128 + tid) * 8;
            *(v8h*)(&sQ[e]) = *(const v8h*)(qsrc + e);
        }
    }
    if (tid < 16) { sRowMax[tid] = -3.0e38f; sRowSum[tid] = 0.0f; }
    __syncthreads();

    Acc acc[8];
#pragma unroll
    for (int t = 0; t < 8; ++t)
#pragma unroll
        for (int r = 0; r < 8; ++r) acc[t].f[r] = 0.0f;

    for (int mb = 0; mb < NMB; ++mb) {
        const int key0 = mb * KEY_BLK;

        // ---- Phase A: raw scores for this megablock -> sP, per-lane max.
        // 16 groups of 64 keys; wave w takes groups w, w+4, w+8, w+12.
        float lmax[8];
#pragma unroll
        for (int r = 0; r < 8; ++r) lmax[r] = -3.0e38f;

        for (int g4 = 0; g4 < 4; ++g4) {
            const int grp  = g4 * 4 + w;               // [0,16): 64-key group
            const int keyb = key0 + grp * 64;          // group's first key
            Acc sc[4];
#pragma unroll
            for (int t = 0; t < 4; ++t)
#pragma unroll
                for (int r = 0; r < 8; ++r) sc[t].f[r] = 0.0f;

            for (int kc = 0; kc < H / 32; ++kc) {
                FragH a;
                const int k0 = kc * 32 + (hi ? 8 : 0);
                a.h[0] = *(const v8h*)(&sQ[ln16 * H + k0]);
                a.h[1] = *(const v8h*)(&sQ[ln16 * H + k0 + 16]);
                // B = K^T: lane holds one key row, K-dim contiguous.
                // 4 tiles share a base pointer; tile stride 16*H fits IOFFSET.
                const _Float16* kp0 =
                    Kh + (size_t)(b * S + keyb + ln16) * H + kc * 32 + (hi ? 8 : 0);
#pragma unroll
                for (int t = 0; t < 4; ++t) {
                    const _Float16* kp = kp0 + (size_t)t * 16 * H;
                    FragH bf;
                    bf.h[0] = *(const v8h*)(kp);
                    bf.h[1] = *(const v8h*)(kp + 16);
                    sc[t].v = wmma_f16(a.v, bf.v, sc[t].v);
                }
            }
#pragma unroll
            for (int t = 0; t < 4; ++t) {
#pragma unroll
                for (int r = 0; r < 8; ++r) {
                    const int m   = r + hi * 8;
                    const int col = grp * 64 + t * 16 + ln16;
                    sP[m * KEY_BLK + col] = (_Float16)sc[t].f[r];
                    lmax[r] = fmaxf(lmax[r], sc[t].f[r]);
                }
            }
        }
#pragma unroll
        for (int r = 0; r < 8; ++r) sPMax[tid][r] = lmax[r];
        __syncthreads();

        // ---- Row max + rescale factor (threads 0..15, one per row)
        if (tid < 16) {
            const int m = tid;
            float om = sRowMax[m];
            float nm = om;
            for (int t = 0; t < 128; ++t)
                if (((t >> 4) & 1) == (m >> 3))
                    nm = fmaxf(nm, sPMax[t][m & 7]);
            sFactor[m] = __expf(om - nm);   // 0 on first block
            sRowMax[m] = nm;
        }
        __syncthreads();

        // ---- Phase B: exponentiate in place, partial row sums
        {
            const int m  = tid >> 3;
            const int c0 = (tid & 7) * 128;
            const float nmax = sRowMax[m];
            float s = 0.0f;
            for (int j = 0; j < 128; ++j) {
                float e = __expf((float)sP[m * KEY_BLK + c0 + j] - nmax);
                sP[m * KEY_BLK + c0 + j] = (_Float16)e;
                s += e;
            }
            sPSum[tid] = s;
        }
        __syncthreads();
        if (tid < 16) {
            float s = 0.0f;
#pragma unroll
            for (int j = 0; j < 8; ++j) s += sPSum[tid * 8 + j];
            sRowSum[tid] = sRowSum[tid] * sFactor[tid] + s;
        }
        __syncthreads();

        // ---- Phase C: rescale accumulators, then P @ V for this megablock
        float fm[8];
#pragma unroll
        for (int r = 0; r < 8; ++r) fm[r] = sFactor[r + hi * 8];
#pragma unroll
        for (int t = 0; t < 8; ++t)
#pragma unroll
            for (int r = 0; r < 8; ++r) acc[t].f[r] *= fm[r];

        for (int kc = 0; kc < KEY_BLK / 32; ++kc) {
            FragH a;
            const int k0 = kc * 32 + (hi ? 8 : 0);
            a.h[0] = *(const v8h*)(&sP[ln16 * KEY_BLK + k0]);
            a.h[1] = *(const v8h*)(&sP[ln16 * KEY_BLK + k0 + 16]);
            const int kb = key0 + kc * 32 + (hi ? 8 : 0);
            // B = V: Vt[b][col][key], key-dim contiguous; col stride S fits IOFFSET
            const _Float16* vp0 = Vt + ((size_t)b * H + w * 128 + ln16) * S + kb;
#pragma unroll
            for (int nt = 0; nt < 8; ++nt) {
                const _Float16* vp = vp0 + (size_t)nt * 16 * S;
                FragH bf;
                bf.h[0] = *(const v8h*)(vp);
                bf.h[1] = *(const v8h*)(vp + 16);
                acc[nt].v = wmma_f16(a.v, bf.v, acc[nt].v);
            }
        }
        __syncthreads();   // sP reused next megablock; sRowSum final after last
    }

    // ---- Epilogue: normalize and store fp32 output
#pragma unroll
    for (int r = 0; r < 8; ++r) {
        const int m = r + hi * 8;
        const float inv = 1.0f / sRowSum[m];
#pragma unroll
        for (int nt = 0; nt < 8; ++nt) {
            const int col = w * 128 + nt * 16 + ln16;
            out[(size_t)(b * S + q0 + m) * H + col] = acc[nt].f[r] * inv;
        }
    }
}

// ---------------------------------------------------------------------------
extern "C" void kernel_launch(void* const* d_in, const int* in_sizes, int n_in,
                              void* d_out, int out_size, void* d_ws, size_t ws_size,
                              hipStream_t stream) {
    (void)in_sizes; (void)n_in; (void)out_size; (void)ws_size;
    const float* X  = (const float*)d_in[0];
    const float* Wq = (const float*)d_in[1];
    const float* bq = (const float*)d_in[2];
    const float* Wk = (const float*)d_in[3];
    const float* bk = (const float*)d_in[4];
    const float* Wv = (const float*)d_in[5];
    const float* bv = (const float*)d_in[6];
    float* out = (float*)d_out;

    char* ws = (char*)d_ws;
    _Float16* Qh = (_Float16*)(ws);                               // 16 MB
    _Float16* Kh = (_Float16*)(ws + (size_t)MT * H * 2);          // 16 MB
    _Float16* Vt = (_Float16*)(ws + (size_t)MT * H * 4);          // 16 MB
    _Float16* Wt = (_Float16*)(ws + (size_t)MT * H * 6);          // 1.5 MB

    wt_kernel<<<(3 * H * H) / 256, 256, 0, stream>>>(Wq, Wk, Wv, Wt);
    proj_kernel<<<3072, 256, 0, stream>>>(X, Wt, bq, bk, bv, Qh, Kh, Vt);
    attn_kernel<<<B * (S / 16), 128, 0, stream>>>(Qh, Kh, Vt, out);
}
